// CausalSelfAttention_44521630990365
// MI455X (gfx1250) — matrix operand order repaired and checked by reading, not compiled, not run on hardware
//
#include <hip/hip_runtime.h>

typedef __bf16 bf16;
typedef unsigned int u32;
typedef unsigned long long u64;
typedef __attribute__((ext_vector_type(4)))  __bf16 v4bf;
typedef __attribute__((ext_vector_type(8)))  __bf16 v8bf;
typedef __attribute__((ext_vector_type(16))) __bf16 v16bf;
typedef __attribute__((ext_vector_type(8)))  float  v8f;
typedef __attribute__((ext_vector_type(4)))  u32    v4u32;
typedef __attribute__((ext_vector_type(8)))  int    v8i32;
typedef __attribute__((ext_vector_type(4)))  int    v4i32;

#define WMMA_BF16(a, b, c) \
  __builtin_amdgcn_wmma_f32_16x16x32_bf16(false, (a), false, (b), (short)0, (c), false, false)

constexpr int Bb = 2, Tt = 4096, Cc = 768, NH = 12, HD = 64;
constexpr int Mrows = Bb * Tt;                       // 8192
constexpr size_t HEADSZ = (size_t)Bb * NH * Tt * HD; // 6291456 elems per Q/K/V

// ---------------------------------------------------------------------------
// helpers
// ---------------------------------------------------------------------------
__device__ inline v16bf mk16(v8bf lo, v8bf hi) {
  v16bf r;
#pragma unroll
  for (int i = 0; i < 8; ++i) { r[i] = lo[i]; r[i + 8] = hi[i]; }
  return r;
}

__device__ inline float redmax16(float x) {
#pragma unroll
  for (int m = 1; m <= 8; m <<= 1) x = fmaxf(x, __shfl_xor(x, m, 32));
  return x;
}
__device__ inline float redsum16(float x) {
#pragma unroll
  for (int m = 1; m <= 8; m <<= 1) x += __shfl_xor(x, m, 32);
  return x;
}

__device__ inline u32 lds_addr_of(const void* p) {  // low 32 bits of generic = LDS byte addr
  return (u32)(u64)(size_t)p;
}

// ---------------------------------------------------------------------------
// Tensor Data Mover: 2D tile load Global -> LDS, bf16 (data_size=2B), with
// LDS row padding. D# layout per CDNA5 ISA 8.3/8.4.
//   pad_int: code for DWORDs-per-row (3 => 16 dw = 64B row, 4 => 32 dw = 128B)
//   pad_amt: code for pad DWORDs minus 1 (3 => 4 dw = 16B, 7 => 8 dw = 32B)
// This toolchain exposes the 6-arg builtin:
//   (uint32x4 g0, int32x8 g1, int32x4 g2, int32x4 g3, int32x8 g4, i32 cpol)
// ---------------------------------------------------------------------------
__device__ inline void tdm_load_2d(u32 lds_addr, u64 gaddr,
                                   u32 tile_d0, u32 tile_d1, u64 stride0,
                                   u32 pad_int, u32 pad_amt) {
  const u32 td0 = 1u << 20, td1 = 1u << 20;  // tensor dims (OOB never hit; tiles in-bounds)
  v4u32 g0;
  g0[0] = 1u;                                             // count=1, user descriptor
  g0[1] = lds_addr;                                       // LDS byte address
  g0[2] = (u32)gaddr;                                     // global addr lo
  g0[3] = ((u32)(gaddr >> 32) & 0x01ffffffu) | (2u << 30);// global addr hi | type=2
  v8i32 g1;
  g1[0] = (int)((1u << 16) | (1u << 20) | (pad_int << 22) | (pad_amt << 25)); // 2B, pad_en
  g1[1] = (int)((td0 & 0xffffu) << 16);                   // abar_addr=0 | tensor_dim0 lo16
  g1[2] = (int)((td0 >> 16) | ((td1 & 0xffffu) << 16));   // tensor_dim0 hi | tensor_dim1 lo
  g1[3] = (int)((td1 >> 16) | (tile_d0 << 16));           // tensor_dim1 hi | tile_dim0
  g1[4] = (int)(tile_d1 & 0xffffu);                       // tile_dim1 | tile_dim2=0
  g1[5] = (int)(u32)stride0;                              // dim0 stride lo32 (elements)
  g1[6] = (int)((u32)(stride0 >> 32) & 0xffffu);          // dim0 stride hi | dim1 stride=0
  g1[7] = 0;
  v4i32 gz4 = {0, 0, 0, 0};                               // 2D: groups 2/3 zero (unused dims)
  v8i32 gz8 = {0, 0, 0, 0, 0, 0, 0, 0};
  __builtin_amdgcn_tensor_load_to_lds(g0, g1, gz4, gz4, gz8, 0);
}

// ---------------------------------------------------------------------------
// fp32 -> bf16 pack (vectorized)
// ---------------------------------------------------------------------------
__global__ void pack_bf16(const float* __restrict__ src, bf16* __restrict__ dst, int n4) {
  int i = blockIdx.x * blockDim.x + threadIdx.x;
  if (i >= n4) return;
  float4 f = ((const float4*)src)[i];
  v4bf o;
  o[0] = (bf16)f.x; o[1] = (bf16)f.y; o[2] = (bf16)f.z; o[3] = (bf16)f.w;
  ((v4bf*)dst)[i] = o;
}

// fp32 [K][N] -> bf16 [N][K] (one-time weight transpose; coalesced reads)
__global__ void pack_transpose_bf16(const float* __restrict__ src, bf16* __restrict__ dst,
                                    int K, int N) {
  int idx = blockIdx.x * blockDim.x + threadIdx.x;
  if (idx >= K * N) return;
  int k = idx / N, n = idx % N;
  dst[(size_t)n * K + k] = (bf16)src[idx];
}

// ---------------------------------------------------------------------------
// Tiled bf16 GEMM with WMMA + double-buffered TDM tile loads.
//   out = A[MxK] * BT[NxK]^T + bias[N]
// Block tile 128x128, K-step 32; 8 waves (4x2), each wave 32x64 (2x4 WMMAs).
// MODE 0: bias + scatter bf16 into head-major Q/K (and transposed V).
// MODE 1: bias + fp32 store to d_out.
// ---------------------------------------------------------------------------
template <int MODE>
__global__ __launch_bounds__(256) void gemm_bf16_wmma(
    const bf16* __restrict__ A, const bf16* __restrict__ BT,
    const float* __restrict__ bias, void* __restrict__ outp,
    int M, int N, int K) {
  __shared__ bf16 As[2][128][40];  // [m][k], stride 80B; TDM pads 16B per 64B row
  __shared__ bf16 Bs[2][128][40];  // [n][k]

  const int tid = threadIdx.x;
  const int lane = tid & 31, wave = tid >> 5;
  const int wm = wave & 3, wn = wave >> 2;
  const int row = lane & 15, hf = lane >> 4;
  const int mbase = blockIdx.x * 128, nbase = blockIdx.y * 128;

  const v8f zero8 = {0.f, 0.f, 0.f, 0.f, 0.f, 0.f, 0.f, 0.f};
  v8f acc[2][4];
#pragma unroll
  for (int mt = 0; mt < 2; ++mt)
#pragma unroll
    for (int nt = 0; nt < 4; ++nt) acc[mt][nt] = zero8;

  const int niter = K / 32;
  auto issue = [&](int it, int buf) {
    const int k0 = it * 32;
    tdm_load_2d(lds_addr_of(&As[buf][0][0]), (u64)(size_t)(A + (size_t)mbase * K + k0),
                32, 128, (u64)K, 3, 3);
    tdm_load_2d(lds_addr_of(&Bs[buf][0][0]), (u64)(size_t)(BT + (size_t)nbase * K + k0),
                32, 128, (u64)K, 3, 3);
  };
  if (wave == 0) issue(0, 0);

  for (int it = 0; it < niter; ++it) {
    const int buf = it & 1;
    __syncthreads();  // all readers done with buf^1 before TDM overwrites it
    if (wave == 0) {
      if (it + 1 < niter) {
        issue(it + 1, buf ^ 1);
        __builtin_amdgcn_s_wait_tensorcnt(2);  // in-order => tile `it` resident
      } else {
        __builtin_amdgcn_s_wait_tensorcnt(0);
      }
    }
    __syncthreads();  // broadcast tile readiness

    v16bf afrag[2], bfrag[4];
#pragma unroll
    for (int mt = 0; mt < 2; ++mt) {  // A layout: K = hf*16 + e (contiguous)
      const bf16* p = &As[buf][wm * 32 + mt * 16 + row][0];
      afrag[mt] = mk16(*(const v8bf*)(p + hf * 16), *(const v8bf*)(p + hf * 16 + 8));
    }
#pragma unroll
    for (int nt = 0; nt < 4; ++nt) {  // B layout: K = hf*16 + e (contiguous)
      const bf16* p = &Bs[buf][wn * 64 + nt * 16 + row][hf * 16];
      bfrag[nt] = mk16(*(const v8bf*)p, *(const v8bf*)(p + 8));
    }
#pragma unroll
    for (int mt = 0; mt < 2; ++mt)
#pragma unroll
      for (int nt = 0; nt < 4; ++nt)
        acc[mt][nt] = WMMA_BF16(afrag[mt], bfrag[nt], acc[mt][nt]);
  }

  // epilogue: C/D layout N = lane%16, M = hf*8 + vgpr
#pragma unroll
  for (int mt = 0; mt < 2; ++mt)
#pragma unroll
    for (int nt = 0; nt < 4; ++nt)
#pragma unroll
      for (int v = 0; v < 8; ++v) {
        int gm = mbase + wm * 32 + mt * 16 + hf * 8 + v;
        int gn = nbase + wn * 64 + nt * 16 + row;
        float val = acc[mt][nt][v] + bias[gn];
        if (MODE == 0) {
          int which = gn / Cc, cc = gn % Cc;
          int h = cc >> 6, d = cc & 63;
          int b = gm / Tt, t = gm % Tt;
          size_t base = (size_t)which * HEADSZ;
          if (which == 2)  // V stored transposed: [B,NH,HD,T] for TDM-friendly P*V tiles
            ((bf16*)outp)[base + (((size_t)(b * NH + h) * HD + d) * Tt + t)] = (bf16)val;
          else             // Q,K head-major: [B,NH,T,HD]
            ((bf16*)outp)[base + (((size_t)(b * NH + h) * Tt + t) * HD + d)] = (bf16)val;
        } else {
          ((float*)outp)[(size_t)gm * N + gn] = val;
        }
      }
}

// ---------------------------------------------------------------------------
// Flash attention, causal. Grid: (B*NH, T/128). 8 waves x 16 q rows.
// K/V tiles double-buffered via TDM; V pre-transposed in global memory.
// ---------------------------------------------------------------------------
__global__ __launch_bounds__(256) void flash_attn_wmma(
    const bf16* __restrict__ qb, const bf16* __restrict__ kb,
    const bf16* __restrict__ vt, bf16* __restrict__ yb) {
  __shared__ bf16 Ks[2][64][80];   // [key][d]  (B operand of Q*K^T)
  __shared__ bf16 Vt[2][64][80];   // [d][key]  (B operand of P*V)
  __shared__ bf16 Ps[8][16][80];   // per-wave P scratch: C/D layout -> A layout

  const int tid = threadIdx.x, lane = tid & 31, wave = tid >> 5;
  const int row = lane & 15, hf = lane >> 4;
  const int bh = blockIdx.x;
  const int qbase = blockIdx.y * 128;
  const int qrow0 = qbase + wave * 16;
  const size_t hoff = (size_t)bh * Tt * HD;  // same for head-major K and transposed V

  // Q fragments in registers for the whole kernel (A operand, 2 K-steps)
  v16bf qf[2];
#pragma unroll
  for (int s = 0; s < 2; ++s) {
    const bf16* p = qb + hoff + (size_t)(qrow0 + row) * HD + s * 32;
    qf[s] = mk16(*(const v8bf*)(p + hf * 16), *(const v8bf*)(p + hf * 16 + 8));
  }

  const v8f zero8 = {0.f, 0.f, 0.f, 0.f, 0.f, 0.f, 0.f, 0.f};
  v8f oacc[4];
#pragma unroll
  for (int c = 0; c < 4; ++c) oacc[c] = zero8;
  float mrow[8], lrow[8];
#pragma unroll
  for (int v = 0; v < 8; ++v) { mrow[v] = -3.0e38f; lrow[v] = 0.f; }

  const int nkb = (qbase >> 6) + 2;  // key blocks with kbase < qbase+128 (causal)
  auto issue = [&](int kbi, int buf) {
    const int kbase = kbi * 64;
    tdm_load_2d(lds_addr_of(&Ks[buf][0][0]),
                (u64)(size_t)(kb + hoff + (size_t)kbase * HD), 64, 64, (u64)HD, 4, 7);
    tdm_load_2d(lds_addr_of(&Vt[buf][0][0]),
                (u64)(size_t)(vt + hoff + kbase), 64, 64, (u64)Tt, 4, 7);
  };
  if (wave == 0) issue(0, 0);

  for (int kbi = 0; kbi < nkb; ++kbi) {
    const int kbase = kbi * 64;
    const int buf = kbi & 1;
    __syncthreads();  // readers done with buf^1
    if (wave == 0) {
      if (kbi + 1 < nkb) {
        issue(kbi + 1, buf ^ 1);
        __builtin_amdgcn_s_wait_tensorcnt(2);
      } else {
        __builtin_amdgcn_s_wait_tensorcnt(0);
      }
    }
    __syncthreads();

    // S = Q * K^T (16 x 64 per wave)
    v8f sacc[4];
#pragma unroll
    for (int c = 0; c < 4; ++c) sacc[c] = zero8;
#pragma unroll
    for (int ks = 0; ks < 2; ++ks) {
#pragma unroll
      for (int c = 0; c < 4; ++c) {
        const bf16* p = &Ks[buf][c * 16 + row][ks * 32 + hf * 16];
        v16bf bfr = mk16(*(const v8bf*)p, *(const v8bf*)(p + 8));
        sacc[c] = WMMA_BF16(qf[ks], bfr, sacc[c]);
      }
    }

    // scale + causal mask
    const bool crossing = (kbase + 63 > qrow0);
#pragma unroll
    for (int c = 0; c < 4; ++c)
#pragma unroll
      for (int v = 0; v < 8; ++v) {
        float x = sacc[c][v] * 0.125f;  // 1/sqrt(64)
        if (crossing && (kbase + c * 16 + row > qrow0 + hf * 8 + v)) x = -3.0e38f;
        sacc[c][v] = x;
      }

    // online softmax (row reductions across 16-lane half-groups)
#pragma unroll
    for (int v = 0; v < 8; ++v) {
      float t = fmaxf(fmaxf(sacc[0][v], sacc[1][v]), fmaxf(sacc[2][v], sacc[3][v]));
      t = redmax16(t);
      float mn = fmaxf(mrow[v], t);
      float al = __expf(mrow[v] - mn);
      float rs = 0.f;
#pragma unroll
      for (int c = 0; c < 4; ++c) {
        float e_ = __expf(sacc[c][v] - mn);
        sacc[c][v] = e_;
        rs += e_;
      }
      rs = redsum16(rs);
      lrow[v] = lrow[v] * al + rs;
      mrow[v] = mn;
#pragma unroll
      for (int c = 0; c < 4; ++c) oacc[c][v] *= al;
    }

    // P: C/D layout -> bf16 A layout via wave-private LDS scratch
#pragma unroll
    for (int c = 0; c < 4; ++c)
#pragma unroll
      for (int v = 0; v < 8; ++v)
        Ps[wave][hf * 8 + v][c * 16 + row] = (bf16)sacc[c][v];

    // O += P * V (contraction over 64 keys, 2 K-steps)
#pragma unroll
    for (int ks = 0; ks < 2; ++ks) {
      const bf16* pp = &Ps[wave][row][ks * 32];
      v16bf pa = mk16(*(const v8bf*)(pp + hf * 16), *(const v8bf*)(pp + hf * 16 + 8));
#pragma unroll
      for (int c = 0; c < 4; ++c) {
        const bf16* vp = &Vt[buf][c * 16 + row][ks * 32 + hf * 16];
        v16bf bfr = mk16(*(const v8bf*)vp, *(const v8bf*)(vp + 8));
        oacc[c] = WMMA_BF16(pa, bfr, oacc[c]);
      }
    }
  }

  // normalize and scatter into [B, T, C] bf16 for the projection GEMM
  const int b = bh / NH, h = bh % NH;
#pragma unroll
  for (int v = 0; v < 8; ++v) {
    const float inv = 1.0f / lrow[v];
    const int q = qrow0 + hf * 8 + v;
#pragma unroll
    for (int c = 0; c < 4; ++c) {
      int d = c * 16 + row;
      yb[((size_t)(b * Tt + q)) * Cc + h * HD + d] = (bf16)(oacc[c][v] * inv);
    }
  }
}

// ---------------------------------------------------------------------------
// launch
// ---------------------------------------------------------------------------
extern "C" void kernel_launch(void* const* d_in, const int* in_sizes, int n_in,
                              void* d_out, int out_size, void* d_ws, size_t ws_size,
                              hipStream_t stream) {
  const float* x      = (const float*)d_in[0];
  const float* w_attn = (const float*)d_in[1];
  const float* b_attn = (const float*)d_in[2];
  const float* w_proj = (const float*)d_in[3];
  const float* b_proj = (const float*)d_in[4];
  float* out = (float*)d_out;

  // workspace carve-up (all offsets multiples of 256B)
  char* ws = (char*)d_ws;
  bf16* xb   = (bf16*)(ws);                                             // 8192x768
  bf16* waT  = (bf16*)(ws + 12582912);                                  // [2304][768]
  bf16* wpT  = (bf16*)(ws + 12582912 + 3538944);                        // [768][768]
  bf16* qkvb = (bf16*)(ws + 12582912 + 3538944 + 1179648);              // Q,K head-major; V transposed
  bf16* yb   = (bf16*)(ws + 12582912 + 3538944 + 1179648 + 37748736);   // 8192x768

  // packs / weight transposes (one-time)
  pack_bf16<<<dim3((Mrows * Cc / 4 + 255) / 256), 256, 0, stream>>>(x, xb, Mrows * Cc / 4);
  pack_transpose_bf16<<<dim3((Cc * 3 * Cc + 255) / 256), 256, 0, stream>>>(w_attn, waT, Cc, 3 * Cc);
  pack_transpose_bf16<<<dim3((Cc * Cc + 255) / 256), 256, 0, stream>>>(w_proj, wpT, Cc, Cc);

  // QKV projection: 8192 x 2304 x 768
  gemm_bf16_wmma<0><<<dim3(Mrows / 128, (3 * Cc) / 128), 256, 0, stream>>>(
      xb, waT, b_attn, (void*)qkvb, Mrows, 3 * Cc, Cc);

  const bf16* qb = qkvb;
  const bf16* kb = qkvb + HEADSZ;
  const bf16* vt = qkvb + 2 * HEADSZ;
  flash_attn_wmma<<<dim3(Bb * NH, Tt / 128), 256, 0, stream>>>(qb, kb, vt, yb);

  // output projection: 8192 x 768 x 768 -> fp32 d_out
  gemm_bf16_wmma<1><<<dim3(Mrows / 128, Cc / 128), 256, 0, stream>>>(
      yb, wpT, b_proj, (void*)out, Mrows, Cc, Cc);

  (void)in_sizes; (void)n_in; (void)out_size; (void)ws_size;
}
